// MultiheadSelfAttentionDec_807453852396
// MI455X (gfx1250) — compile-verified, matrix operand order checked by reading, not run on hardware
//
#include <hip/hip_runtime.h>

// ---------------------------------------------------------------------------
// MultiheadSelfAttention for MI455X (gfx1250).
// bf16 WMMA for all matmuls + Tensor Data Mover (TDM) double-buffered LDS
// staging. S=4096, C=2048, H=16, D=128.
// ---------------------------------------------------------------------------

constexpr int S_ = 4096;
constexpr int C_ = 2048;
constexpr int H_ = 16;
constexpr int D_ = 128;
constexpr float EPS_ = 1.1920929e-07f;

typedef __bf16 bf16_t;
typedef bf16_t v16bf __attribute__((ext_vector_type(16)));
typedef float  v8f   __attribute__((ext_vector_type(8)));
typedef unsigned int v4u __attribute__((ext_vector_type(4)));
typedef unsigned int v8u __attribute__((ext_vector_type(8)));
typedef unsigned short u16;

__device__ __forceinline__ u16 f2bf(float f) {
  unsigned int u = __float_as_uint(f);
  u += 0x7FFFu + ((u >> 16) & 1u);   // round-to-nearest-even
  return (u16)(u >> 16);
}

// Load one 16x32 bf16 A/B fragment for v_wmma_f32_16x16x32_bf16.
// lanes 0-15 hold K={0..7,16..23}, lanes 16-31 hold K={8..15,24..31}.
__device__ __forceinline__ v16bf ldfrag(const u16* p) {
  v16bf f;
  reinterpret_cast<uint4*>(&f)[0] = *reinterpret_cast<const uint4*>(p);
  reinterpret_cast<uint4*>(&f)[1] = *reinterpret_cast<const uint4*>(p + 16);
  return f;
}

__device__ __forceinline__ v8f wmma_bf16(v16bf a, v16bf b, v8f c) {
  return __builtin_amdgcn_wmma_f32_16x16x32_bf16(false, a, false, b,
                                                 (short)0, c, false, false);
}

// ---------------------------------------------------------------------------
// TDM: async 2-D tile DMA global -> LDS (2-byte elements), ISA ch.8 D#.
// Tile window == tensor window (d0 x d1), dim0 contiguous, row stride
// `stride_elems`. LDS destination is compacted row-major (d0 per row).
// Tracked by TENSORcnt; pair with __builtin_amdgcn_s_wait_tensorcnt(0).
// ---------------------------------------------------------------------------
__device__ __forceinline__ void tdm_load_2d(unsigned lds_off, const u16* gptr,
                                            unsigned d0, unsigned d1,
                                            unsigned long long stride_elems) {
  unsigned long long ga = (unsigned long long)(uintptr_t)gptr;
  v4u g0;
  g0[0] = 1u;                                           // count=1, user mode
  g0[1] = lds_off;                                      // lds_addr (bytes)
  g0[2] = (unsigned)ga;                                 // global_addr[31:0]
  g0[3] = (unsigned)((ga >> 32) & 0x01FFFFFFu) | (2u << 30);  // addr[56:32]|type=2
  v8u g1;
  g1[0] = 1u << 16;                                     // data_size=1 (2 bytes)
  g1[1] = (d0 & 0xFFFFu) << 16;                         // tensor_dim0 lo
  g1[2] = ((d0 >> 16) & 0xFFFFu) | ((d1 & 0xFFFFu) << 16);   // dim0 hi | dim1 lo
  g1[3] = ((d1 >> 16) & 0xFFFFu) | ((d0 & 0xFFFFu) << 16);   // dim1 hi | tile_dim0
  g1[4] = (d1 & 0xFFFFu);                               // tile_dim1 (tile_dim2=0)
  g1[5] = (unsigned)(stride_elems & 0xFFFFFFFFu);       // tensor_dim0_stride lo
  g1[6] = (unsigned)((stride_elems >> 32) & 0xFFFFu);   // stride hi (dim1_stride=0)
  g1[7] = 0u;
  v4u gz = {0u, 0u, 0u, 0u};
  asm volatile("tensor_load_to_lds %0, %1, %2, %3"
               :: "s"(g0), "s"(g1), "s"(gz), "s"(gz)
               : "memory");
}

// ---------------------------------------------------------------------------
// f32 -> bf16 conversion
// ---------------------------------------------------------------------------
__global__ __launch_bounds__(256)
void cvt_bf16_kernel(const float* __restrict__ in, u16* __restrict__ out, int n) {
  int i = (blockIdx.x * blockDim.x + threadIdx.x) * 4;
  if (i + 3 < n) {
    float4 v = *reinterpret_cast<const float4*>(in + i);
    ushort4 o;
    o.x = f2bf(v.x); o.y = f2bf(v.y); o.z = f2bf(v.z); o.w = f2bf(v.w);
    *reinterpret_cast<ushort4*>(out + i) = o;
  }
}

// ---------------------------------------------------------------------------
// Tiled WMMA GEMM: out[m,n] = sum_k A[m,k] * W[n,k]  (+ epilogue)
// Block = 128 threads (4 waves). Tile = 128x128, K-step 32.
// TDM double-buffered A/B tiles: wave0 DMAs A, wave1 DMAs B.
// MODE 0: q  -> RMSNorm(qn_w) + rotary, bf16 out [S,C]
// MODE 1: k  -> RMSNorm(kn_w) + rotary, bf16 out [S,C]
// MODE 2: v  -> bias only, bf16 out transposed per head: vT[(h*D+d)*S + s]
// MODE 3: o-proj -> bias, f32 out [S,C]
// ---------------------------------------------------------------------------
union GemmSmem {
  struct { u16 A[2][128 * 32]; u16 B[2][128 * 32]; } t;   // 4 x 8KB
  float stage[128 * 128];                                 // 64KB epilogue stage
};
// LDS byte offsets (single shared object, allocated at LDS offset 0):
constexpr unsigned GEMM_A_OFF = 0;
constexpr unsigned GEMM_B_OFF = 16384;

template <int MODE>
__global__ __launch_bounds__(128)
void gemm128_kernel(const u16* __restrict__ A, const u16* __restrict__ W,
                    const float* __restrict__ bias, const float* __restrict__ nw,
                    const float* __restrict__ rope, u16* __restrict__ outb,
                    float* __restrict__ outf) {
  __shared__ GemmSmem sm;

  const int tid  = threadIdx.x;
  const int lane = tid & 31;
  const int wv   = tid >> 5;
  const int wvu  = __builtin_amdgcn_readfirstlane(wv);   // scalar wave id
  const int wm   = (wv >> 1) * 64;
  const int wn   = (wv & 1) * 64;
  const int m0   = blockIdx.x * 128;
  const int n0   = blockIdx.y * 128;
  const int l15  = lane & 15;
  const int kb0  = (lane >> 4) * 8;   // K sub-chunk base per half-wave
  const int roff = (lane >> 4) * 8;   // row offset of C layout

  v8f acc[4][4];
#pragma unroll
  for (int i = 0; i < 4; i++)
#pragma unroll
    for (int j = 0; j < 4; j++)
      acc[i][j] = v8f{0.f, 0.f, 0.f, 0.f, 0.f, 0.f, 0.f, 0.f};

  // Prologue: DMA the first K tiles into buffer 0.
  if (wvu == 0)
    tdm_load_2d(GEMM_A_OFF, A + (size_t)m0 * C_, 32, 128, C_);
  else if (wvu == 1)
    tdm_load_2d(GEMM_B_OFF, W + (size_t)n0 * C_, 32, 128, C_);

  for (int k0 = 0; k0 < C_; k0 += 32) {
    const int ib = (k0 >> 5) & 1;
    __builtin_amdgcn_s_wait_tensorcnt(0);
    __syncthreads();
    if (k0 + 32 < C_) {
      const int nb = ib ^ 1;
      if (wvu == 0)
        tdm_load_2d(GEMM_A_OFF + nb * 8192, A + (size_t)m0 * C_ + k0 + 32,
                    32, 128, C_);
      else if (wvu == 1)
        tdm_load_2d(GEMM_B_OFF + nb * 8192, W + (size_t)n0 * C_ + k0 + 32,
                    32, 128, C_);
    }

    const u16* At = sm.t.A[ib];
    const u16* Bt = sm.t.B[ib];
    v16bf af[4], bf[4];
#pragma unroll
    for (int i = 0; i < 4; i++)
      af[i] = ldfrag(At + (wm + i * 16 + l15) * 32 + kb0);
#pragma unroll
    for (int j = 0; j < 4; j++)
      bf[j] = ldfrag(Bt + (wn + j * 16 + l15) * 32 + kb0);

#pragma unroll
    for (int i = 0; i < 4; i++)
#pragma unroll
      for (int j = 0; j < 4; j++)
        acc[i][j] = wmma_bf16(af[i], bf[j], acc[i][j]);
  }

  // Spill accumulators (C layout: VGPR r -> row r / r+8 per half-wave) to LDS.
  __syncthreads();
#pragma unroll
  for (int i = 0; i < 4; i++)
#pragma unroll
    for (int j = 0; j < 4; j++)
#pragma unroll
      for (int r = 0; r < 8; r++)
        sm.stage[(wm + i * 16 + roff + r) * 128 + (wn + j * 16 + l15)] = acc[i][j][r];
  __syncthreads();

  // Per-row epilogue: thread t owns full row t (128 cols == one head dim).
  const int s = m0 + tid;
  float* row = &sm.stage[tid * 128];

  if (MODE <= 1) {
    float ss = 0.f;
    for (int c = 0; c < 128; c++) {
      float t = row[c] + bias[n0 + c];
      row[c] = t;
      ss += t * t;
    }
    const float inv = rsqrtf(ss * (1.0f / 128.0f) + EPS_);
    const float* rp = rope + (size_t)s * 256;   // [64][2][2]
    u16* orow = outb + (size_t)s * C_ + n0;
    for (int i = 0; i < 64; i++) {
      float x0 = row[2 * i + 0] * inv * nw[2 * i + 0];
      float x1 = row[2 * i + 1] * inv * nw[2 * i + 1];
      float o0 = rp[i * 4 + 0] * x0 + rp[i * 4 + 1] * x1;
      float o1 = rp[i * 4 + 2] * x0 + rp[i * 4 + 3] * x1;
      orow[2 * i + 0] = f2bf(o0);
      orow[2 * i + 1] = f2bf(o1);
    }
  } else if (MODE == 2) {
    for (int c = 0; c < 128; c++)
      outb[(size_t)(n0 + c) * S_ + s] = f2bf(row[c] + bias[n0 + c]);
  } else {
    for (int c = 0; c < 128; c++)
      outf[(size_t)s * C_ + n0 + c] = row[c] + bias[n0 + c];
  }
}

// ---------------------------------------------------------------------------
// Flash attention: block = 256 threads (8 waves), grid = (S/128, H).
// Wave w owns q rows [blk*128 + w*16, +16). KV blocks of 128 are TDM
// double-buffered in LDS: wave0 DMAs K rows, wave1 DMAs V^T rows.
// ---------------------------------------------------------------------------
struct AttnSmem {
  u16 Ks[2][128 * 128];    // 2 x 32KB, K rows x d
  u16 VTs[2][128 * 128];   // 2 x 32KB, d rows x kv cols
  u16 Ps[8][16 * 128];     // 32KB, per-wave P tile (A-frag staging)
};
constexpr unsigned ATT_KS_OFF = 0;
constexpr unsigned ATT_VT_OFF = 65536;

__global__ __launch_bounds__(256)
void attention_kernel(const u16* __restrict__ qb, const u16* __restrict__ kb,
                      const u16* __restrict__ vT, u16* __restrict__ attn) {
  __shared__ AttnSmem sm;

  const int tid  = threadIdx.x;
  const int lane = tid & 31;
  const int wv   = tid >> 5;
  const int wvu  = __builtin_amdgcn_readfirstlane(wv);
  const int h    = blockIdx.y;
  const int q0   = blockIdx.x * 128 + wv * 16;
  const int l15  = lane & 15;
  const int kb0  = (lane >> 4) * 8;
  const int roff = (lane >> 4) * 8;
  const float scale = 0.08838834764831845f;   // 1/sqrt(128)

  // Q fragments held in registers for the whole KV sweep (16 x 128 per wave).
  v16bf qf[4];
#pragma unroll
  for (int ks = 0; ks < 4; ks++)
    qf[ks] = ldfrag(qb + (size_t)(q0 + l15) * C_ + h * D_ + ks * 32 + kb0);

  v8f oacc[8];
#pragma unroll
  for (int j = 0; j < 8; j++)
    oacc[j] = v8f{0.f, 0.f, 0.f, 0.f, 0.f, 0.f, 0.f, 0.f};
  float m_i[8], l_i[8];
#pragma unroll
  for (int r = 0; r < 8; r++) { m_i[r] = -1e30f; l_i[r] = 0.f; }

  // Prologue: DMA KV block 0 into buffer 0.
  if (wvu == 0)
    tdm_load_2d(ATT_KS_OFF, kb + (size_t)0 * C_ + h * D_, 128, 128, C_);
  else if (wvu == 1)
    tdm_load_2d(ATT_VT_OFF, vT + (size_t)(h * D_) * S_, 128, 128, S_);

  for (int kv = 0; kv < S_; kv += 128) {
    const int buf = (kv >> 7) & 1;
    __builtin_amdgcn_s_wait_tensorcnt(0);
    __syncthreads();
    if (kv + 128 < S_) {
      const int nb = buf ^ 1;
      if (wvu == 0)
        tdm_load_2d(ATT_KS_OFF + nb * 32768,
                    kb + (size_t)(kv + 128) * C_ + h * D_, 128, 128, C_);
      else if (wvu == 1)
        tdm_load_2d(ATT_VT_OFF + nb * 32768,
                    vT + (size_t)(h * D_) * S_ + kv + 128, 128, 128, S_);
    }
    const u16* Ks  = sm.Ks[buf];
    const u16* VTs = sm.VTs[buf];

    // S = q . k^T  (16 x 128 per wave): 8 N-tiles x 4 K-steps of WMMA.
    v8f sacc[8];
#pragma unroll
    for (int j = 0; j < 8; j++) {
      sacc[j] = v8f{0.f, 0.f, 0.f, 0.f, 0.f, 0.f, 0.f, 0.f};
#pragma unroll
      for (int ks = 0; ks < 4; ks++) {
        v16bf kf = ldfrag(Ks + (j * 16 + l15) * 128 + ks * 32 + kb0);
        sacc[j] = wmma_bf16(qf[ks], kf, sacc[j]);
      }
    }

    // Online softmax; acc slot r is one row; cols live across 16 lanes.
#pragma unroll
    for (int r = 0; r < 8; r++) {
      float tm = -1e30f;
#pragma unroll
      for (int j = 0; j < 8; j++) tm = fmaxf(tm, sacc[j][r]);
      tm *= scale;
#pragma unroll
      for (int off = 1; off < 16; off <<= 1)
        tm = fmaxf(tm, __shfl_xor(tm, off, 16));
      float mnew  = fmaxf(m_i[r], tm);
      float alpha = __expf(m_i[r] - mnew);
      float ls = 0.f;
#pragma unroll
      for (int j = 0; j < 8; j++) {
        float p = __expf(sacc[j][r] * scale - mnew);
        sacc[j][r] = p;
        ls += p;
      }
#pragma unroll
      for (int off = 1; off < 16; off <<= 1)
        ls += __shfl_xor(ls, off, 16);
      l_i[r] = l_i[r] * alpha + ls;
      m_i[r] = mnew;
#pragma unroll
      for (int jd = 0; jd < 8; jd++) oacc[jd][r] *= alpha;
    }

    // Re-lay P from C-layout into A-fragment layout via per-wave LDS tile.
#pragma unroll
    for (int j = 0; j < 8; j++)
#pragma unroll
      for (int r = 0; r < 8; r++)
        sm.Ps[wv][(roff + r) * 128 + j * 16 + l15] = f2bf(sacc[j][r]);
    __asm__ volatile("s_wait_dscnt 0" ::: "memory");  // in-wave LDS RAW fence

    // O += P . V : A = P (16 x 128), B = vT rows (d cols contiguous).
#pragma unroll
    for (int ks = 0; ks < 4; ks++) {
      v16bf pf = ldfrag(&sm.Ps[wv][l15 * 128 + ks * 32 + kb0]);
#pragma unroll
      for (int jd = 0; jd < 8; jd++) {
        v16bf vf = ldfrag(VTs + (jd * 16 + l15) * 128 + ks * 32 + kb0);
        oacc[jd] = wmma_bf16(pf, vf, oacc[jd]);
      }
    }
  }

  // Finalize: out = O / l, bf16 into attn[s, h*D + d].
#pragma unroll
  for (int jd = 0; jd < 8; jd++)
#pragma unroll
    for (int r = 0; r < 8; r++)
      attn[(size_t)(q0 + roff + r) * C_ + h * D_ + jd * 16 + l15] =
          f2bf(oacc[jd][r] / l_i[r]);
}

// ---------------------------------------------------------------------------
// Launcher. Workspace layout (bytes):
//  [0,16M)   xb      bf16 x            [16M,24M)  Wqb   [24M,32M) Wkb
//  [32M,40M) Wvb     [40M,48M) Wob
//  [48M,64M) qb      [64M,80M) kbuf    [80M,96M)  vT    [96M,112M) attn
// ---------------------------------------------------------------------------
extern "C" void kernel_launch(void* const* d_in, const int* in_sizes, int n_in,
                              void* d_out, int out_size, void* d_ws, size_t ws_size,
                              hipStream_t stream) {
  (void)in_sizes; (void)n_in; (void)out_size; (void)ws_size;
  const float* x    = (const float*)d_in[0];
  const float* rope = (const float*)d_in[1];
  const float* Wq   = (const float*)d_in[2];
  const float* bq   = (const float*)d_in[3];
  const float* Wk   = (const float*)d_in[4];
  const float* bk   = (const float*)d_in[5];
  const float* Wv   = (const float*)d_in[6];
  const float* bv   = (const float*)d_in[7];
  const float* qn_w = (const float*)d_in[8];
  const float* kn_w = (const float*)d_in[9];
  const float* Wo   = (const float*)d_in[10];
  const float* bo   = (const float*)d_in[11];
  float* out = (float*)d_out;

  char* ws = (char*)d_ws;
  const size_t MB = 1u << 20;
  u16* xb   = (u16*)(ws + 0 * MB);
  u16* Wqb  = (u16*)(ws + 16 * MB);
  u16* Wkb  = (u16*)(ws + 24 * MB);
  u16* Wvb  = (u16*)(ws + 32 * MB);
  u16* Wob  = (u16*)(ws + 40 * MB);
  u16* qb   = (u16*)(ws + 48 * MB);
  u16* kbuf = (u16*)(ws + 64 * MB);
  u16* vT   = (u16*)(ws + 80 * MB);
  u16* ab   = (u16*)(ws + 96 * MB);

  auto cvt = [&](const float* i, u16* o, int n) {
    cvt_bf16_kernel<<<(n / 4 + 255) / 256, 256, 0, stream>>>(i, o, n);
  };
  cvt(x, xb, S_ * C_);
  cvt(Wq, Wqb, C_ * C_);
  cvt(Wk, Wkb, C_ * C_);
  cvt(Wv, Wvb, C_ * C_);
  cvt(Wo, Wob, C_ * C_);

  dim3 g(S_ / 128, C_ / 128);
  gemm128_kernel<0><<<g, 128, 0, stream>>>(xb, Wqb, bq, qn_w, rope, qb, nullptr);
  gemm128_kernel<1><<<g, 128, 0, stream>>>(xb, Wkb, bk, kn_w, rope, kbuf, nullptr);
  gemm128_kernel<2><<<g, 128, 0, stream>>>(xb, Wvb, bv, nullptr, nullptr, vT, nullptr);

  attention_kernel<<<dim3(S_ / 128, H_), 256, 0, stream>>>(qb, kbuf, vT, ab);

  gemm128_kernel<3><<<g, 128, 0, stream>>>(ab, Wob, bo, nullptr, nullptr, nullptr, out);
}
